// SingleHead2DAttention_57131654971623
// MI455X (gfx1250) — compile-verified, matrix operand order checked by reading, not run on hardware
//
#include <hip/hip_runtime.h>
#include <hip/hip_bf16.h>
#include <hip/hip_fp16.h>

typedef __attribute__((ext_vector_type(2)))  float    v2f;
typedef __attribute__((ext_vector_type(8)))  float    v8f;
typedef __attribute__((ext_vector_type(16))) _Float16 v16h;

#define B_   2
#define C_   32
#define H_   96
#define W_   128
#define HW_  (H_ * W_)      // 12288
#define BC_  (B_ * C_)      // 64
#define EPSF 1e-6f

// ---- workspace layout (in floats) ----
#define OFF_QS    0                          // [B,C,H]   6144
#define OFF_QC    6144                       // [B,C,W]   8192
#define OFF_KH    14336                      // [B,C,H]   6144
#define OFF_KW    20480                      // [B,C,W]   8192
#define OFF_SQ    28672                      // [B,C]     64
#define OFF_SK    28736                      // [B,C]     64
#define OFF_GMAX  28800                      // [32][4]   128  (atomic max slots)
#define OFF_GSMAX 28928                      // [32]      32
#define OFF_HR    28960                      // [32][B,C,H]
#define OFF_HC    (OFF_HR + 32 * 6144)
#define OFF_WR    (OFF_HC + 32 * 6144)
#define OFF_WC    (OFF_WR + 32 * 8192)
#define OFF_IRS   (OFF_WC + 32 * 8192)       // [4][64][224] inverse row/col sums
#define OFF_SCORE (OFF_IRS + 4 * 64 * 224)   // [B,C,H,W] 786432 (reused per c0)

#if __has_builtin(__builtin_amdgcn_wmma_f32_16x16x4_f32)
#define USE_WMMA_F32 1
#endif

// ---------------- K1: per-(b,c) row sums, col sums, totals of Q and K ------
__global__ void k1_sums(const float* __restrict__ q, const float* __restrict__ k,
                        float* __restrict__ ws) {
  __shared__ float tile[HW_];
  __shared__ float rowbuf[H_];
  const int bc  = blockIdx.x;    // 0..63
  const int tid = threadIdx.x;   // 256
  for (int p = 0; p < 2; ++p) {
    const float* s = p ? k : q;
    for (int i = tid; i < HW_; i += 256) tile[i] = s[bc * HW_ + i];
    __syncthreads();
    if (tid < H_) {
      float acc = 0.f;
      for (int x = 0; x < W_; ++x) acc += tile[tid * W_ + x];
      ws[(p ? OFF_KH : OFF_QS) + bc * H_ + tid] = acc;
      rowbuf[tid] = acc;
    }
    if (tid >= 128) {
      const int x = tid - 128;   // 0..127
      float acc = 0.f;
      for (int h = 0; h < H_; ++h) acc += tile[h * W_ + x];
      ws[(p ? OFF_KW : OFF_QC) + bc * W_ + x] = acc;
    }
    __syncthreads();
    if (tid == 0) {
      float t = 0.f;
      for (int h = 0; h < H_; ++h) t += rowbuf[h];
      ws[(p ? OFF_SK : OFF_SQ) + bc] = t;
    }
    __syncthreads();
  }
}

// ---------------- K2: build + min-max-normalize the 4 vector families ------
// block = (c0, type): type 0=h_row 1=h_col 2=w_row 3=w_col
__global__ void k2_norm(float* __restrict__ ws) {
  const int type = blockIdx.x & 3;
  const int c0   = blockIdx.x >> 2;
  const int tid  = threadIdx.x;
  __shared__ float rmn[256], rmx[256];
  const int   len    = (type < 2) ? H_ : W_;
  const int   N      = BC_ * len;
  const float addeps = (type < 2) ? (H_ * EPSF) : (W_ * EPSF);
  float* dst;
  if      (type == 0) dst = ws + OFF_HR + c0 * (BC_ * H_);
  else if (type == 1) dst = ws + OFF_HC + c0 * (BC_ * H_);
  else if (type == 2) dst = ws + OFF_WR + c0 * (BC_ * W_);
  else                dst = ws + OFF_WC + c0 * (BC_ * W_);
  float lmn = 3.4e38f, lmx = -3.4e38f;
  for (int i = tid; i < N; i += 256) {
    const int b = i / (C_ * len);
    const int r = i % (C_ * len);
    const int c = r / len;
    const int x = r % len;
    float raw;
    if      (type == 0) raw = ws[OFF_SQ + b * C_ + c0] * ws[OFF_KH + (b * C_ + c) * H_ + x] + addeps;
    else if (type == 1) raw = ws[OFF_QS + (b * C_ + c0) * H_ + x] * ws[OFF_SK + b * C_ + c] + addeps;
    else if (type == 2) raw = ws[OFF_SQ + b * C_ + c0] * ws[OFF_KW + (b * C_ + c) * W_ + x] + addeps;
    else                raw = ws[OFF_QC + (b * C_ + c0) * W_ + x] * ws[OFF_SK + b * C_ + c] + addeps;
    dst[i] = raw;
    lmn = fminf(lmn, raw);
    lmx = fmaxf(lmx, raw);
  }
  rmn[tid] = lmn; rmx[tid] = lmx;
  __syncthreads();
  for (int s = 128; s > 0; s >>= 1) {
    if (tid < s) { rmn[tid] = fminf(rmn[tid], rmn[tid + s]); rmx[tid] = fmaxf(rmx[tid], rmx[tid + s]); }
    __syncthreads();
  }
  const float mn  = rmn[0];
  const float inv = 1.0f / (rmx[0] - mn);
  for (int i = tid; i < N; i += 256) dst[i] = 2.0f + (dst[i] - mn) * inv;
  // re-init the atomic-max slots once per call (graph-replay safe)
  if (blockIdx.x == 0)
    for (int i = tid; i < 160; i += 256) ws[OFF_GMAX + i] = 0.0f;
}

// ---------------- K3: per (term, b, c) tile: E, rowsum/colsum via WMMA -----
__global__ void k3_tile(float* __restrict__ ws, int c0) {
  const int t   = blockIdx.x >> 6;   // 0..3
  const int bc  = blockIdx.x & 63;
  const int tid = threadIdx.x;       // 256
  __shared__ float E[HW_];
  __shared__ float mv[H_], nv[W_];
  __shared__ float rsb[H_], csb[W_];
  __shared__ float red[256];

  const float* mptr = ws + ((t < 2) ? OFF_HR : OFF_HC) + c0 * (BC_ * H_) + bc * H_;
  const float* nptr = ws + (((t & 1) == 0) ? OFF_WR : OFF_WC) + c0 * (BC_ * W_) + bc * W_;
  if (tid < H_)   mv[tid] = mptr[tid];
  if (tid >= 128) nv[tid - 128] = nptr[tid - 128];
  __syncthreads();

  for (int i = tid; i < HW_; i += 256) E[i] = __expf(mv[i >> 7] * nv[i & 127]);
  __syncthreads();

  const int wave = tid >> 5, lane = tid & 31;
#if defined(USE_WMMA_F32)
  if (wave < 6) {                                   // row sums: E(16x4) x ones(4x16)
    const int M   = wave * 16 + (lane & 15);
    const int kho = (lane < 16) ? 0 : 2;
    const v2f ones = {1.0f, 1.0f};
    v8f acc = {};
    for (int k0 = 0; k0 < W_; k0 += 4) {
      v2f a = { E[M * W_ + k0 + kho], E[M * W_ + k0 + kho + 1] };
      acc = __builtin_amdgcn_wmma_f32_16x16x4_f32(false, a, false, ones, (short)0, acc, false, false);
    }
    if ((lane & 15) == 0) {
      const int rbase = wave * 16 + ((lane >> 4) << 3);
      #pragma unroll
      for (int v = 0; v < 8; ++v) rsb[rbase + v] = acc[v];
    }
  } else {                                          // col sums: ones(16x4) x E(4x16)
    const int N   = lane & 15;
    const int kho = (lane < 16) ? 0 : 2;
    const v2f onesA = {1.0f, 1.0f};
    for (int nb = (wave - 6) * 4; nb < (wave - 6) * 4 + 4; ++nb) {
      const int N0 = nb * 16;
      v8f acc = {};
      for (int k0 = 0; k0 < H_; k0 += 4) {
        v2f b = { E[(k0 + kho) * W_ + N0 + N], E[(k0 + kho + 1) * W_ + N0 + N] };
        acc = __builtin_amdgcn_wmma_f32_16x16x4_f32(false, onesA, false, b, (short)0, acc, false, false);
      }
      if (lane < 16) csb[N0 + N] = acc[0];
    }
  }
#else  // f16 fallback (codegen-confirmed builtin), same ones-matrix trick
  if (wave < 6) {
    const int M = wave * 16 + (lane & 15);
    const int khalf = (lane < 16) ? 0 : 8;
    v16h hb;
    #pragma unroll
    for (int e = 0; e < 16; ++e) hb[e] = (_Float16)1.0f;
    v8f acc = {};
    for (int k0 = 0; k0 < W_; k0 += 32) {
      v16h ha;
      #pragma unroll
      for (int v = 0; v < 8; ++v) {
        const int kp = (v & 3) * 2 + ((v >> 2) << 4) + khalf;
        ha[2 * v]     = (_Float16)E[M * W_ + k0 + kp];
        ha[2 * v + 1] = (_Float16)E[M * W_ + k0 + kp + 1];
      }
      acc = __builtin_amdgcn_wmma_f32_16x16x32_f16(false, ha, false, hb, (short)0, acc, false, false);
    }
    if ((lane & 15) == 0) {
      const int rbase = wave * 16 + ((lane >> 4) << 3);
      #pragma unroll
      for (int v = 0; v < 8; ++v) rsb[rbase + v] = acc[v];
    }
  } else {
    const int N = lane & 15;
    const int khalf = (lane < 16) ? 0 : 16;
    v16h ha;
    #pragma unroll
    for (int e = 0; e < 16; ++e) ha[e] = (_Float16)1.0f;
    for (int nb = (wave - 6) * 4; nb < (wave - 6) * 4 + 4; ++nb) {
      const int N0 = nb * 16;
      v8f acc = {};
      for (int k0 = 0; k0 < H_; k0 += 32) {
        v16h hbm;
        #pragma unroll
        for (int v = 0; v < 8; ++v) {
          const int kk = khalf + 2 * v;
          hbm[2 * v]     = (_Float16)E[(k0 + kk) * W_ + N0 + N];
          hbm[2 * v + 1] = (_Float16)E[(k0 + kk + 1) * W_ + N0 + N];
        }
        acc = __builtin_amdgcn_wmma_f32_16x16x32_f16(false, ha, false, hbm, (short)0, acc, false, false);
      }
      if (lane < 16) csb[N0 + N] = acc[0];
    }
  }
#endif
  __syncthreads();

  float* irs = ws + OFF_IRS + (t * 64 + bc) * 224;
  if (tid < H_)   { const float v = 1.0f / rsb[tid]; rsb[tid] = v; irs[tid] = v; }
  if (tid >= 128) { const int x = tid - 128; const float v = 1.0f / csb[x]; csb[x] = v; irs[96 + x] = v; }
  __syncthreads();

  // rc = E * (1/rowsum + 1/colsum); tile max -> global per-(c0,term) max
  float lmax = 0.0f;
  for (int i = tid; i < HW_; i += 256)
    lmax = fmaxf(lmax, E[i] * (rsb[i >> 7] + csb[i & 127]));
  red[tid] = lmax;
  __syncthreads();
  for (int s = 128; s > 0; s >>= 1) {
    if (tid < s) red[tid] = fmaxf(red[tid], red[tid + s]);
    __syncthreads();
  }
  if (tid == 0)
    atomicMax((int*)(ws + OFF_GMAX + c0 * 4 + t), __float_as_int(red[0]));
}

// ---------------- K4: score = sum_t rc_t / gmax_t ; track global max -------
__global__ void k4_score(float* __restrict__ ws, int c0) {
  const int bc  = blockIdx.x;    // 0..63
  const int tid = threadIdx.x;
  __shared__ float m2[2][H_], n2[2][W_], ir[4][H_], ic[4][W_], red[256];
  for (int i = tid; i < 2 * H_; i += 256) {
    const int s = i / H_, h = i % H_;
    m2[s][h] = ws[(s ? OFF_HC : OFF_HR) + c0 * (BC_ * H_) + bc * H_ + h];
  }
  for (int i = tid; i < 2 * W_; i += 256) {
    const int s = i / W_, x = i % W_;
    n2[s][x] = ws[(s ? OFF_WC : OFF_WR) + c0 * (BC_ * W_) + bc * W_ + x];
  }
  for (int i = tid; i < 4 * 224; i += 256) {
    const int t = i / 224, r = i % 224;
    const float v = ws[OFF_IRS + (t * 64 + bc) * 224 + r];
    if (r < 96) ir[t][r] = v; else ic[t][r - 96] = v;
  }
  __syncthreads();
  float ig[4];
  #pragma unroll
  for (int t = 0; t < 4; ++t) ig[t] = 1.0f / ws[OFF_GMAX + c0 * 4 + t];

  float* sc = ws + OFF_SCORE + bc * HW_;
  float lmax = 0.0f;
  for (int i = tid; i < HW_; i += 256) {
    const int h = i >> 7, x = i & 127;
    float s = 0.f;
    #pragma unroll
    for (int t = 0; t < 4; ++t) {
      const float e = __expf(m2[t >> 1][h] * n2[t & 1][x]);
      s += e * (ir[t][h] + ic[t][x]) * ig[t];
    }
    sc[i] = s;
    lmax = fmaxf(lmax, s);
  }
  red[tid] = lmax;
  __syncthreads();
  for (int s2 = 128; s2 > 0; s2 >>= 1) {
    if (tid < s2) red[tid] = fmaxf(red[tid], red[tid + s2]);
    __syncthreads();
  }
  if (tid == 0)
    atomicMax((int*)(ws + OFF_GSMAX + c0), __float_as_int(red[0]));
}

// ---------------- K5: z[:,c0] = sum_c (score/gsmax) * values ---------------
__global__ void k5_z(const float* __restrict__ ws, const float* __restrict__ values,
                     float* __restrict__ out, int c0) {
  const int gid = blockIdx.x * 256 + threadIdx.x;   // < B*H*W = 24576
  const int b = gid / HW_;
  const int r = gid % HW_;
  const float inv = 1.0f / ws[OFF_GSMAX + c0];
  float z = 0.f;
  for (int c = 0; c < C_; ++c) {
    const int off = (b * C_ + c) * HW_ + r;
    z += ws[OFF_SCORE + off] * values[off];
  }
  out[(b * C_ + c0) * HW_ + r] = z * inv;
}

// ---------------- K6: emit normalized score of last channel ----------------
__global__ void k6_scoreout(const float* __restrict__ ws, float* __restrict__ out) {
  const int gid = blockIdx.x * 256 + threadIdx.x;   // < 786432
  out[BC_ * HW_ + gid] = ws[OFF_SCORE + gid] * (1.0f / ws[OFF_GSMAX + 31]);
}

extern "C" void kernel_launch(void* const* d_in, const int* in_sizes, int n_in,
                              void* d_out, int out_size, void* d_ws, size_t ws_size,
                              hipStream_t stream) {
  (void)in_sizes; (void)n_in; (void)out_size; (void)ws_size;
  const float* q   = (const float*)d_in[0];
  const float* k   = (const float*)d_in[1];
  const float* val = (const float*)d_in[2];
  float* out = (float*)d_out;
  float* ws  = (float*)d_ws;

  k1_sums<<<64, 256, 0, stream>>>(q, k, ws);
  k2_norm<<<128, 256, 0, stream>>>(ws);
  for (int c0 = 0; c0 < C_; ++c0) {
    k3_tile<<<256, 256, 0, stream>>>(ws, c0);     // 4 terms x 64 (b,c) tiles
    k4_score<<<64, 256, 0, stream>>>(ws, c0);
    k5_z<<<96, 256, 0, stream>>>(ws, val, out, c0);
  }
  k6_scoreout<<<3072, 256, 0, stream>>>(ws, out);
}
